// SimilarityPoolingBatch_15075335208983
// MI455X (gfx1250) — matrix operand order repaired and checked by reading, not compiled, not run on hardware
//
#include <hip/hip_runtime.h>

typedef __attribute__((ext_vector_type(16))) _Float16 v16h;
typedef __attribute__((ext_vector_type(8)))  float    v8f;

#define BB 16
#define TT 4096
#define CC 256
#define THRESH 0.9f
#define EPSF 1e-12f

// ---------------------------------------------------------------------------
// Kernel 1 (fused norms + sims): one wave per 16-pair tile.
//   A = frames t0..t0+15 (rows), B = frames t0+1..t0+16 (cols).
//   Diagonal of D = A*B (f16 in, f32 accum, K looped over C=256) = dot(t,t+1).
//   Norms are computed in f32 from the same loads: lanes L and L^16 hold
//   complementary K-halves of the same row for both A and B operands, so one
//   shfl_xor(…,16) yields ||row||^2 exactly — no separate norm pass needed.
// ---------------------------------------------------------------------------
__global__ __launch_bounds__(256) void k_sim(const float* __restrict__ x,
                                             const int*   __restrict__ mask,
                                             int*         __restrict__ starts) {
  int wave = (blockIdx.x * 256 + threadIdx.x) >> 5;   // [0, BB * TT/16)
  int lane = threadIdx.x & 31;
  int b    = wave >> 8;          // TT/16 = 256 tiles per row
  int t0   = (wave & 255) << 4;
  int half = lane >> 4;          // 0: lanes 0-15, 1: lanes 16-31
  int l16  = lane & 15;

  // A per-lane: row M = l16; K chunks [koff..koff+7],[koff+16..koff+23],
  // koff = half ? 8 : 0   (ISA 16-bit A 16x32 layout).
  int rowA = t0 + l16;
  // B per-lane: column N = l16 (frame t0+1+N); K chunks [koff..koff+7],
  // [koff+16..koff+23], koff = half ? 8 : 0.  Clamp last tile's t+16 row (result unused).
  int rowB = min(t0 + 1 + l16, TT - 1);
  const float* pa = x + (size_t)(b * TT + rowA) * CC + (half ? 8 : 0);
  const float* pb = x + (size_t)(b * TT + rowB) * CC + (half ? 8 : 0);

  v8f acc = {};
  float nA = 0.0f, nB = 0.0f;   // f32 sum-of-squares (half of row each)
#pragma unroll
  for (int kk = 0; kk < CC; kk += 32) {
    __builtin_prefetch(pa + kk + 32, 0, 1);   // global_prefetch_b8
    float4 a0 = ((const float4*)(pa + kk))[0];
    float4 a1 = ((const float4*)(pa + kk))[1];
    float4 a2 = ((const float4*)(pa + kk + 16))[0];
    float4 a3 = ((const float4*)(pa + kk + 16))[1];
    float4 b0 = ((const float4*)(pb + kk))[0];
    float4 b1 = ((const float4*)(pb + kk))[1];
    float4 b2 = ((const float4*)(pb + kk + 16))[0];
    float4 b3 = ((const float4*)(pb + kk + 16))[1];
    nA += a0.x*a0.x + a0.y*a0.y + a0.z*a0.z + a0.w*a0.w
        + a1.x*a1.x + a1.y*a1.y + a1.z*a1.z + a1.w*a1.w
        + a2.x*a2.x + a2.y*a2.y + a2.z*a2.z + a2.w*a2.w
        + a3.x*a3.x + a3.y*a3.y + a3.z*a3.z + a3.w*a3.w;
    nB += b0.x*b0.x + b0.y*b0.y + b0.z*b0.z + b0.w*b0.w
        + b1.x*b1.x + b1.y*b1.y + b1.z*b1.z + b1.w*b1.w
        + b2.x*b2.x + b2.y*b2.y + b2.z*b2.z + b2.w*b2.w
        + b3.x*b3.x + b3.y*b3.y + b3.z*b3.z + b3.w*b3.w;
    v16h av, bv;
    av[0] = (_Float16)a0.x; av[1] = (_Float16)a0.y; av[2]  = (_Float16)a0.z; av[3]  = (_Float16)a0.w;
    av[4] = (_Float16)a1.x; av[5] = (_Float16)a1.y; av[6]  = (_Float16)a1.z; av[7]  = (_Float16)a1.w;
    av[8] = (_Float16)a2.x; av[9] = (_Float16)a2.y; av[10] = (_Float16)a2.z; av[11] = (_Float16)a2.w;
    av[12]= (_Float16)a3.x; av[13]= (_Float16)a3.y; av[14] = (_Float16)a3.z; av[15] = (_Float16)a3.w;
    bv[0] = (_Float16)b0.x; bv[1] = (_Float16)b0.y; bv[2]  = (_Float16)b0.z; bv[3]  = (_Float16)b0.w;
    bv[4] = (_Float16)b1.x; bv[5] = (_Float16)b1.y; bv[6]  = (_Float16)b1.z; bv[7]  = (_Float16)b1.w;
    bv[8] = (_Float16)b2.x; bv[9] = (_Float16)b2.y; bv[10] = (_Float16)b2.z; bv[11] = (_Float16)b2.w;
    bv[12]= (_Float16)b3.x; bv[13]= (_Float16)b3.y; bv[14] = (_Float16)b3.z; bv[15] = (_Float16)b3.w;
    acc = __builtin_amdgcn_wmma_f32_16x16x32_f16(false, av, false, bv,
                                                 (short)0, acc, false, false);
  }

  // Complete the row norms: lane L and lane L^16 hold complementary K-halves.
  nA += __shfl_xor(nA, 16, 32);             // ||x[t0+l16]||^2
  nB += __shfl_xor(nB, 16, 32);             // ||x[t0+1+l16]||^2
  float rnA = 1.0f / fmaxf(sqrtf(nA), EPSF);
  float rnB = 1.0f / fmaxf(sqrtf(nB), EPSF);

  // Diagonal extraction: C/D layout — VGPR r, lanes 0-15 -> (M=r, N=lane);
  // lanes 16-31 -> (M=8+r, N=lane-16).  Diagonal (m,m) is held by the lane
  // with l16 == m, whose own rnA/rnB are exactly norms of frames m and m+1.
#pragma unroll
  for (int r = 0; r < 8; ++r) {
    int m = half ? (r + 8) : r;
    if (m == l16) {
      int t = t0 + m, tn = t + 1;
      if (tn < TT) {
        float sim = acc[r] * rnA * rnB;
        int vp = (mask[b * TT + t] == 0) && (mask[b * TT + tn] == 0);
        starts[b * TT + tn] = (sim < THRESH && vp) ? 1 : 0;
      }
    }
  }
}

// ---------------------------------------------------------------------------
// Kernel 2: per-row inclusive scan of starts -> seg ids.  One block per row.
// ---------------------------------------------------------------------------
__global__ __launch_bounds__(256) void k_scan(const int* __restrict__ mask,
                                              const int* __restrict__ starts,
                                              int*       __restrict__ segids) {
  const int PER = TT / 256;  // 16 elements per thread
  int b   = blockIdx.x;
  int tid = threadIdx.x;
  int base = b * TT + tid * PER;

  int vals[PER];
  int tsum = 0;
#pragma unroll
  for (int i = 0; i < PER; ++i) {
    int v;
    if (tid == 0 && i == 0) v = (mask[b * TT] == 0) ? 1 : 0;  // starts[b,0]
    else                    v = starts[base + i];
    vals[i] = v;
    tsum += v;
  }

  __shared__ int sh[256];
  sh[tid] = tsum;
  __syncthreads();
#pragma unroll
  for (int off = 1; off < 256; off <<= 1) {
    int v = (tid >= off) ? sh[tid - off] : 0;
    __syncthreads();
    sh[tid] += v;
    __syncthreads();
  }
  int run = sh[tid] - tsum;  // exclusive prefix of this thread's chunk
#pragma unroll
  for (int i = 0; i < PER; ++i) {
    run += vals[i];
    int idx = base + i;
    segids[idx] = (mask[idx] == 1) ? -1 : (run - 1);
  }
}

// ---------------------------------------------------------------------------
// Kernel 3: zero pooled output + counts (re-done every call: graph replays).
// ---------------------------------------------------------------------------
__global__ void k_zero(float* __restrict__ pooled, int* __restrict__ counts) {
  size_t n1 = (size_t)BB * TT * CC;
  size_t n2 = (size_t)BB * TT;
  size_t stride = (size_t)gridDim.x * blockDim.x;
  for (size_t i = (size_t)blockIdx.x * blockDim.x + threadIdx.x;
       i < n1 + n2; i += stride) {
    if (i < n1) pooled[i] = 0.0f;
    else        counts[i - n1] = 0;
  }
}

// ---------------------------------------------------------------------------
// Kernel 4: scatter-accumulate frames into their segment.  One wave per frame.
// Segment id is wave-uniform; f32 global atomics, near-zero contention
// (random data makes nearly every valid pair a boundary -> tiny segments).
// ---------------------------------------------------------------------------
__global__ __launch_bounds__(256) void k_accum(const float* __restrict__ x,
                                               const int*   __restrict__ segids,
                                               float*       __restrict__ pooled,
                                               int*         __restrict__ counts) {
  int wave = (blockIdx.x * 256 + threadIdx.x) >> 5;  // frame index b*TT + t
  int lane = threadIdx.x & 31;
  int seg = segids[wave];
  if (seg < 0) return;                                // wave-uniform exit
  int b = wave / TT;
  const float* src = x + (size_t)wave * CC + lane * 8;
  float* dst = pooled + (size_t)(b * TT + seg) * CC + lane * 8;
  float4 a = ((const float4*)src)[0];
  float4 c = ((const float4*)src)[1];
  __hip_atomic_fetch_add(dst + 0, a.x, __ATOMIC_RELAXED, __HIP_MEMORY_SCOPE_AGENT);
  __hip_atomic_fetch_add(dst + 1, a.y, __ATOMIC_RELAXED, __HIP_MEMORY_SCOPE_AGENT);
  __hip_atomic_fetch_add(dst + 2, a.z, __ATOMIC_RELAXED, __HIP_MEMORY_SCOPE_AGENT);
  __hip_atomic_fetch_add(dst + 3, a.w, __ATOMIC_RELAXED, __HIP_MEMORY_SCOPE_AGENT);
  __hip_atomic_fetch_add(dst + 4, c.x, __ATOMIC_RELAXED, __HIP_MEMORY_SCOPE_AGENT);
  __hip_atomic_fetch_add(dst + 5, c.y, __ATOMIC_RELAXED, __HIP_MEMORY_SCOPE_AGENT);
  __hip_atomic_fetch_add(dst + 6, c.z, __ATOMIC_RELAXED, __HIP_MEMORY_SCOPE_AGENT);
  __hip_atomic_fetch_add(dst + 7, c.w, __ATOMIC_RELAXED, __HIP_MEMORY_SCOPE_AGENT);
  if (lane == 0)
    __hip_atomic_fetch_add(&counts[b * TT + seg], 1, __ATOMIC_RELAXED,
                           __HIP_MEMORY_SCOPE_AGENT);
}

// ---------------------------------------------------------------------------
// Kernel 5: pooled /= max(count,1); durations, new_mask.  One wave per slot.
// ---------------------------------------------------------------------------
__global__ __launch_bounds__(256) void k_final(float* __restrict__ pooled,
                                               const int* __restrict__ counts,
                                               float* __restrict__ dur,
                                               float* __restrict__ nmask) {
  int wave = (blockIdx.x * 256 + threadIdx.x) >> 5;  // slot index b*TT + s
  int lane = threadIdx.x & 31;
  int cnt = counts[wave];
  float scale = 1.0f / (float)max(cnt, 1);
  float* p = pooled + (size_t)wave * CC + lane * 8;
  float4 a = ((float4*)p)[0];
  float4 b = ((float4*)p)[1];
  a.x *= scale; a.y *= scale; a.z *= scale; a.w *= scale;
  b.x *= scale; b.y *= scale; b.z *= scale; b.w *= scale;
  ((float4*)p)[0] = a;
  ((float4*)p)[1] = b;
  if (lane == 0) {
    dur[wave]   = (float)cnt;
    nmask[wave] = (cnt == 0) ? 1.0f : 0.0f;
  }
}

// ---------------------------------------------------------------------------
extern "C" void kernel_launch(void* const* d_in, const int* in_sizes, int n_in,
                              void* d_out, int out_size, void* d_ws, size_t ws_size,
                              hipStream_t stream) {
  (void)in_sizes; (void)n_in; (void)out_size; (void)ws_size;
  const float* x    = (const float*)d_in[0];   // [BB, TT, CC] f32
  const int*   mask = (const int*)d_in[1];     // [BB, TT] i32

  float* out    = (float*)d_out;
  float* pooled = out;                                   // BB*TT*CC
  float* dur    = out + (size_t)BB * TT * CC;            // BB*TT
  float* nmask  = dur + (size_t)BB * TT;                 // BB*TT

  int* starts = (int*)d_ws;                              // BB*TT i32
  int* segids = starts + (size_t)BB * TT;                // BB*TT i32
  int* counts = segids + (size_t)BB * TT;                // BB*TT i32

  k_sim  <<<BB * (TT / 16) / 8, 256, 0, stream>>>(x, mask, starts);
  k_scan <<<BB,                 256, 0, stream>>>(mask, starts, segids);
  k_zero <<<4096,               256, 0, stream>>>(pooled, counts);
  k_accum<<<BB * TT / 8,        256, 0, stream>>>(x, segids, pooled, counts);
  k_final<<<BB * TT / 8,        256, 0, stream>>>(pooled, counts, dur, nmask);
}
